// InferenceModel_1743756722641
// MI455X (gfx1250) — matrix operand order repaired and checked by reading, not compiled, not run on hardware
//
#include <hip/hip_runtime.h>

typedef __attribute__((ext_vector_type(16))) _Float16 v16h;
typedef __attribute__((ext_vector_type(8)))  _Float16 v8h;
typedef __attribute__((ext_vector_type(8)))  float    v8f;

#define B_SZ   256
#define N_SZ   100
#define K_REAL 10000
#define K_PAD  10016
#define NSTEPS 313      // K_PAD / 32
#define HW     16384
#define BN     64       // columns per workgroup
#define BSTR   40       // LDS column stride in halves (padded from 32)

// Swizzled A storage: fragment-major so GEMM A-loads are fully coalesced.
// Element (m_tile, k_step, lane, e) where e indexes the 16 halves of the
// per-lane A fragment (ISA 7.12.2 16-bit A 16x32 layout).
__device__ __forceinline__ int a_swz(int mt, int s, int lane, int e) {
    return ((mt * NSTEPS + s) * 32 + lane) * 16 + e;
}

// --------------------------------------------------------------------------
// Kernel 1: row softmax of cost = -0.1*latent - 1e5*I, written as f16 in
// WMMA A-fragment layout. One wave per (b, i) row of 100 elements.
// --------------------------------------------------------------------------
__global__ void __launch_bounds__(256) softmax_to_frag(
        const float* __restrict__ latent, _Float16* __restrict__ Ah) {
    const int wave = threadIdx.x >> 5;
    const int lane = threadIdx.x & 31;
    const int row  = blockIdx.x * 8 + wave;          // [0, B*N)
    const int b = row / N_SZ, i = row % N_SZ;
    const float* src = latent + (size_t)b * N_SZ * N_SZ + (size_t)i * N_SZ;

    float c[4];
    float m = -1e30f;
#pragma unroll
    for (int t = 0; t < 4; ++t) {
        int j = lane + 32 * t;
        if (j < N_SZ) {
            float x = -0.1f * src[j];
            if (j == i) x -= 100000.0f;
            c[t] = x;
            m = fmaxf(m, x);
        } else {
            c[t] = -1e30f;
        }
    }
#pragma unroll
    for (int off = 16; off >= 1; off >>= 1) m = fmaxf(m, __shfl_xor(m, off, 32));

    float e4[4];
    float sum = 0.0f;
#pragma unroll
    for (int t = 0; t < 4; ++t) {
        int j = lane + 32 * t;
        float e = (j < N_SZ) ? __expf(c[t] - m) : 0.0f;
        e4[t] = e;
        sum += e;
    }
#pragma unroll
    for (int off = 16; off >= 1; off >>= 1) sum += __shfl_xor(sum, off, 32);
    const float inv = 1.0f / sum;

    const int mt = b >> 4, mrow = b & 15;
#pragma unroll
    for (int t = 0; t < 4; ++t) {
        int j = lane + 32 * t;
        if (j < N_SZ) {
            int k    = i * N_SZ + j;
            int s    = k >> 5, kl = k & 31;
            int half = (kl >> 3) & 1;
            int ln   = half * 16 + mrow;
            int e    = (kl < 16) ? (kl - 8 * half) : (8 + (kl - 16) - 8 * half);
            Ah[a_swz(mt, s, ln, e)] = (_Float16)(e4[t] * inv);
        }
    }
}

// Zero the K padding region [10000, 10016): step 312, k_local 16..31.
__global__ void pad_frag(_Float16* __restrict__ Ah) {
    const int b = threadIdx.x;            // one thread per batch row
    const int mt = b >> 4, mrow = b & 15;
#pragma unroll
    for (int kl = 16; kl < 32; ++kl) {
        int half = (kl >> 3) & 1;
        int ln   = half * 16 + mrow;
        int e    = 8 + (kl - 16) - 8 * half;
        Ah[a_swz(mt, NSTEPS - 1, ln, e)] = (_Float16)0.0f;
    }
}

// --------------------------------------------------------------------------
// Kernel 2: C(256 x 16384) = A(256 x 10016) * E(10016 x 16384), f16 WMMA,
// f32 accumulate, fused 2*(x-0.5) + query-mask epilogue.
// Grid: 256 blocks (one 64-column slab each), 256 threads = 8 waves.
// Wave w: N-tile = w&3 (16 cols), M-tiles (w>>2)*8 .. +7 (8 x 16 rows).
// --------------------------------------------------------------------------
__global__ void __launch_bounds__(256) edge_gemm(
        const _Float16* __restrict__ Ah,
        const float*    __restrict__ edges,
        const float*    __restrict__ qimg,
        float*          __restrict__ out) {
    __shared__ _Float16 lds_b[BN * BSTR];

    const int tid     = threadIdx.x;
    const int wave    = tid >> 5, lane = tid & 31;
    const int half    = lane >> 4, lcol = lane & 15;
    const int n_tile  = wave & 3;
    const int m_group = wave >> 2;
    const int n_base  = blockIdx.x * BN;
    const int n_col   = n_base + n_tile * 16 + lcol;

    const float q = qimg[n_col];                 // fixed per lane

    v8f acc[8];
#pragma unroll
    for (int r = 0; r < 8; ++r) acc[r] = (v8f){0, 0, 0, 0, 0, 0, 0, 0};

    // Cooperative edge-tile staging: thread t loads row k_local, 8 columns.
    const int kl_st = tid >> 3;                  // 0..31
    const int n_off = (tid & 7) * 8;             // 0..56

    for (int s = 0; s < NSTEPS; ++s) {
        const int k0  = s * 32;
        int row = k0 + kl_st;
        if (row > K_REAL - 1) row = K_REAL - 1;  // padded K: weight is zero
        const float4* p = (const float4*)(edges + (size_t)row * HW + n_base + n_off);
        const float4 f0 = p[0];
        const float4 f1 = p[1];

        if (s + 1 < NSTEPS) {                    // prefetch next K-step rows
            int nrow = k0 + 32 + kl_st;
            if (nrow > K_REAL - 1) nrow = K_REAL - 1;
            __builtin_prefetch(edges + (size_t)nrow * HW + n_base + n_off, 0, 1);
        }

        _Float16 h[8];
        h[0] = (_Float16)f0.x; h[1] = (_Float16)f0.y;
        h[2] = (_Float16)f0.z; h[3] = (_Float16)f0.w;
        h[4] = (_Float16)f1.x; h[5] = (_Float16)f1.y;
        h[6] = (_Float16)f1.z; h[7] = (_Float16)f1.w;

        __syncthreads();                         // prior readers done
#pragma unroll
        for (int i2 = 0; i2 < 8; ++i2)
            lds_b[(n_off + i2) * BSTR + kl_st] = h[i2];
        __syncthreads();                         // tile visible

        // B fragment: lane = column n_tile*16+lcol, halves e<8 -> K = 8*half+e, e>=8 -> K = 16+8*half+(e-8)
        const int boff = (n_tile * 16 + lcol) * BSTR + 8 * half;
        const v8h b0 = *(const v8h*)(&lds_b[boff]);
        const v8h b1 = *(const v8h*)(&lds_b[boff + 16]);
        v16h bf;
#pragma unroll
        for (int i2 = 0; i2 < 8; ++i2) { bf[i2] = b0[i2]; bf[i2 + 8] = b1[i2]; }

#pragma unroll
        for (int r = 0; r < 8; ++r) {
            const int mt = m_group * 8 + r;
            const v16h af = *(const v16h*)(Ah + a_swz(mt, s, lane, 0));
            acc[r] = __builtin_amdgcn_wmma_f32_16x16x32_f16(
                false, af, false, bf, (short)0, acc[r], false, false);
        }
    }

    // Epilogue: img = 2*(acc - 0.5); masked to 1.0 where query == 1.0
#pragma unroll
    for (int r = 0; r < 8; ++r) {
        const int m0 = (m_group * 8 + r) * 16 + 8 * half;   // + v = row M
#pragma unroll
        for (int v = 0; v < 8; ++v) {
            float val = 2.0f * (acc[r][v] - 0.5f);
            if (q == 1.0f) val = 1.0f;
            out[(size_t)(m0 + v) * HW + n_col] = val;
        }
    }
}

extern "C" void kernel_launch(void* const* d_in, const int* in_sizes, int n_in,
                              void* d_out, int out_size, void* d_ws, size_t ws_size,
                              hipStream_t stream) {
    (void)in_sizes; (void)n_in; (void)out_size; (void)ws_size;
    const float* latent = (const float*)d_in[0];   // (256,100,100) f32
    const float* edges  = (const float*)d_in[1];   // (100,100,128,128) f32
    const float* qimg   = (const float*)d_in[2];   // (1,1,128,128) f32
    float* out = (float*)d_out;                    // (256,1,128,128) f32
    _Float16* Ah = (_Float16*)d_ws;                // 16*313*32*16 halves ~ 5.13 MB

    softmax_to_frag<<<(B_SZ * N_SZ) / 8, 256, 0, stream>>>(latent, Ah);
    pad_frag<<<1, 256, 0, stream>>>(Ah);
    edge_gemm<<<HW / BN, 256, 0, stream>>>(Ah, edges, qimg, out);
}
